// TransformerPredictionHead_45311904973223
// MI455X (gfx1250) — matrix operand order repaired and checked by reading, not compiled, not run on hardware
//
#include <hip/hip_runtime.h>

// ---------------------------------------------------------------------------
// Problem constants
// ---------------------------------------------------------------------------
#define BATCH 4
#define CIN   512
#define DM    256
#define NH    8
#define HD    32
#define FF    2048
#define NL    2
#define HW    40
#define SEQ   (HW*HW)          // 1600
#define MROWS (BATCH*SEQ)      // 6400
#define NAC   3
#define NCLS  80

typedef __bf16 bf16_t;
typedef bf16_t v16bf __attribute__((ext_vector_type(16)));
typedef unsigned short ushort8_t __attribute__((ext_vector_type(8)));
typedef float v8f __attribute__((ext_vector_type(8)));

__device__ __forceinline__ unsigned short f2bf(float f) {
    unsigned u = __builtin_bit_cast(unsigned, f);
    unsigned r = u + 0x7FFFu + ((u >> 16) & 1u);   // round-to-nearest-even
    return (unsigned short)(r >> 16);
}

// join two 8x bf16 halves into one 16x bf16 fragment (consecutive VGPRs)
__device__ __forceinline__ v16bf join16(ushort8_t lo, ushort8_t hi) {
    struct U { ushort8_t lo, hi; } u{lo, hi};
    return __builtin_bit_cast(v16bf, u);
}

// Fragment layouts used by this code (contiguous K, not the ISA table order; A and B match):
//  A(16x32,16-bit): lane(M=lane&15, half=lane>>4): elems 0-7  <- K = 16*half+0..7
//                                                  elems 8-15 <- K = 16*half+8..15
//  B(32x16,16-bit): lane(N=lane&15, half=lane>>4): elems 0-15 <- K = 16*half+0..15

// async 16B copy global -> LDS (ASYNCcnt-tracked; ISA §15.18.3 op 98).
// The LDS pointer is passed through ptrtoint so the shared object is captured
// by the asm (alias analysis must then keep the later ds_loads live); the
// generic LDS address's low 32 bits are the wave-relative LDS byte address
// (ISA §10.2 aperture mapping), which is what the async op's VDST expects.
__device__ __forceinline__ void async_g2l_b128(void* lptr, const void* gptr) {
    unsigned ldsByte = (unsigned)(unsigned long long)lptr;
    asm volatile("global_load_async_to_lds_b128 %0, %1, off"
                 :: "v"(ldsByte), "v"((unsigned long long)gptr)
                 : "memory");
}
__device__ __forceinline__ void wait_async0() {
    asm volatile("s_wait_asynccnt 0" ::: "memory");
}

// ---------------------------------------------------------------------------
// Weight conversion: W[K,N] (f32, row-major) -> Wt[N64][K] (bf16, transposed,
// zero-padded to the 64-column tile grid).  Run once per GEMM launch.
// ---------------------------------------------------------------------------
__global__ void convert_weight(const float* __restrict__ W,
                               unsigned short* __restrict__ Wt, int K, int N)
{
    int idx = blockIdx.x * 256 + threadIdx.x;      // over N64*K
    int n = idx / K, k = idx % K;
    float v = (n < N) ? W[(size_t)k * N + n] : 0.0f;
    Wt[idx] = f2bf(v);
}

// ---------------------------------------------------------------------------
// bf16-WMMA GEMM:  C[M,N] = epilogue( A[M,K](bf16) x Wt[N][K](bf16) )
// epilogue: + bias[n] (+ pos[n]) (+ resid[m,n] f32) ; optional ReLU
// outputs: Cf (f32, nullable) and/or Cb (bf16, nullable)
// Block: 128 threads (4 waves).  Tile: 64(M) x 64(N) x 32(K).
// Tiles are filled with GLOBAL_LOAD_ASYNC_TO_LDS_B128 (ASYNCcnt), then
// fragments are two ds_load_b128 each.
// ---------------------------------------------------------------------------
__global__ __launch_bounds__(128)
void gemm_bf16_wmma(const unsigned short* __restrict__ A,
                    const unsigned short* __restrict__ Wt,
                    const float* __restrict__ bias, const float* __restrict__ pos,
                    const float* __restrict__ resid,
                    float* __restrict__ Cf, unsigned short* __restrict__ Cb,
                    int M, int K, int N, int relu)
{
    __shared__ unsigned short As[64 * 32];   // [row][k]
    __shared__ unsigned short Bs[64 * 32];   // [col][k]

    const int tid  = threadIdx.x;
    const int lane = tid & 31;
    const int w    = tid >> 5;
    const int half = (lane >> 4) & 1;
    const int l15  = lane & 15;
    const int m0   = blockIdx.y * 64;
    const int n0   = blockIdx.x * 64;

    const v8f zero8 = {0.f,0.f,0.f,0.f,0.f,0.f,0.f,0.f};
    v8f acc[4];
#pragma unroll
    for (int c = 0; c < 4; ++c) acc[c] = zero8;

    for (int k0 = 0; k0 < K; k0 += 32) {
        // tile fills: async 16-byte copies straight into LDS (2 per tile/thread)
#pragma unroll
        for (int g = 0; g < 2; ++g) {
            int idx = (tid + g * 128) * 8;           // 0..2040, multiple of 8
            int r = idx >> 5, kk = idx & 31;
            async_g2l_b128(&As[idx], A  + (size_t)(m0 + r) * K + k0 + kk);
            async_g2l_b128(&Bs[idx], Wt + (size_t)(n0 + r) * K + k0 + kk);
        }
        wait_async0();
        __syncthreads();

        // fragments: two ds_load_b128 each
        const unsigned short* ar = &As[(w * 16 + l15) * 32];
        v16bf a = join16(*(const ushort8_t*)(ar + half * 16),
                         *(const ushort8_t*)(ar + half * 16 + 8));
#pragma unroll
        for (int c = 0; c < 4; ++c) {
            const unsigned short* br = &Bs[(c * 16 + l15) * 32 + half * 16];
            v16bf b = join16(*(const ushort8_t*)(br),
                             *(const ushort8_t*)(br + 8));
            acc[c] = __builtin_amdgcn_wmma_f32_16x16x32_bf16(
                         false, a, false, b, (short)0, acc[c], false, false);
        }
        __syncthreads();
    }

    // Epilogue.  C/D layout: vgpr r -> row r + 8*half, col = lane&15.
    const int rhalf = half << 3;
#pragma unroll
    for (int c = 0; c < 4; ++c) {
        int col = n0 + c * 16 + l15;
        if (col < N) {
#pragma unroll
            for (int r = 0; r < 8; ++r) {
                int row = m0 + w * 16 + r + rhalf;
                float v = acc[c][r] + bias[col];
                if (pos)   v += pos[col];
                if (resid) v += resid[(size_t)row * N + col];
                if (relu)  v = fmaxf(v, 0.0f);
                if (Cf) Cf[(size_t)row * N + col] = v;
                if (Cb) Cb[(size_t)row * N + col] = f2bf(v);
            }
        }
    }
}

// ---------------------------------------------------------------------------
// Flash attention: one wave per (batch, head, 16-query block).
// Q/K/V/O are bf16 [M][DM].  S = Q K^T via WMMA (K-dim = HD = 32), scaled in
// f32 after the WMMA, online softmax in f32, O += P V via WMMA.
// ---------------------------------------------------------------------------
__global__ __launch_bounds__(32)
void attention_wmma(const unsigned short* __restrict__ Q,
                    const unsigned short* __restrict__ Km,
                    const unsigned short* __restrict__ V,
                    unsigned short* __restrict__ O)
{
    __shared__ unsigned short Qs[16 * 32];   // [query][dim]   (A for S)
    __shared__ unsigned short Ksh[32 * 32];  // [key][dim]     (B for S: K=dim)
    __shared__ unsigned short Vt[32 * 32];   // [dim][key]     (B for PV: K=key)
    __shared__ unsigned short Ps[16 * 32];   // [query][key]   (A for PV)

    const int lane  = threadIdx.x;
    const int half  = (lane >> 4) & 1;
    const int l15   = lane & 15;
    const int qb    = blockIdx.x;            // 0..99
    const int head  = blockIdx.y;            // 0..7
    const int batch = blockIdx.z;            // 0..3
    const int bm    = batch * SEQ + qb * 16;
    const int hd0   = head * HD;
    const float scale = 0.17677669529663687f;   // 1/sqrt(32)

    // Q fill: 16 rows x 4 dim-groups = 64 copies of 16B, 2 per lane
#pragma unroll
    for (int g = 0; g < 2; ++g) {
        int task = lane + g * 32;
        int row = task & 15, dg = task >> 4;
        *(ushort8_t*)(&Qs[row * 32 + dg * 8]) =
            *(const ushort8_t*)(Q + (size_t)(bm + row) * DM + hd0 + dg * 8);
    }
    __syncthreads();

    const unsigned short* qr = &Qs[l15 * 32];
    const v16bf qa = join16(*(const ushort8_t*)(qr + half * 16),
                            *(const ushort8_t*)(qr + half * 16 + 8));

    const v8f zero8 = {0.f,0.f,0.f,0.f,0.f,0.f,0.f,0.f};
    v8f o0 = zero8, o1 = zero8;
    float mrun[8], lrun[8];
#pragma unroll
    for (int r = 0; r < 8; ++r) { mrun[r] = -1e30f; lrun[r] = 0.0f; }

    const int kbase = batch * SEQ;
    const int rhalf = half << 3;

    for (int j = 0; j < SEQ / 32; ++j) {
        const int kb = j * 32;
        // K fill: 32 keys x 4 dim-groups = 128 copies of 16B, 4 per lane
#pragma unroll
        for (int g = 0; g < 4; ++g) {
            int task = lane + g * 32;
            int key = task & 31, dg = task >> 5;
            *(ushort8_t*)(&Ksh[key * 32 + dg * 8]) =
                *(const ushort8_t*)(Km + (size_t)(kbase + kb + key) * DM + hd0 + dg * 8);
        }
        // V fill (transposed): 32 dims x 4 key-groups, gather 8 u16 + 1 b128 store
#pragma unroll
        for (int g = 0; g < 4; ++g) {
            int task = lane + g * 32;
            int dim = task & 31, kg = task >> 5;
            ushort8_t pk;
#pragma unroll
            for (int t = 0; t < 8; ++t)
                pk[t] = V[(size_t)(kbase + kb + kg * 8 + t) * DM + hd0 + dim];
            *(ushort8_t*)(&Vt[dim * 32 + kg * 8]) = pk;
        }
        __syncthreads();

        // S = Q K^T : B-frag n=key, K=dim (contiguous in Ksh rows)
        const unsigned short* k0r = &Ksh[l15 * 32 + half * 16];
        const unsigned short* k1r = &Ksh[(16 + l15) * 32 + half * 16];
        v16bf b0 = join16(*(const ushort8_t*)(k0r), *(const ushort8_t*)(k0r + 8));
        v16bf b1 = join16(*(const ushort8_t*)(k1r), *(const ushort8_t*)(k1r + 8));
        v8f s0 = __builtin_amdgcn_wmma_f32_16x16x32_bf16(
                     false, qa, false, b0, (short)0, zero8, false, false);
        v8f s1 = __builtin_amdgcn_wmma_f32_16x16x32_bf16(
                     false, qa, false, b1, (short)0, zero8, false, false);

        // online softmax per query row (16-lane half-wave reductions)
#pragma unroll
        for (int r = 0; r < 8; ++r) {
            float sv0 = s0[r] * scale;
            float sv1 = s1[r] * scale;
            float mloc = fmaxf(sv0, sv1);
            mloc = fmaxf(mloc, __shfl_xor(mloc, 1, 32));
            mloc = fmaxf(mloc, __shfl_xor(mloc, 2, 32));
            mloc = fmaxf(mloc, __shfl_xor(mloc, 4, 32));
            mloc = fmaxf(mloc, __shfl_xor(mloc, 8, 32));
            float newm = fmaxf(mrun[r], mloc);
            float corr = __expf(mrun[r] - newm);
            float p0   = __expf(sv0 - newm);
            float p1   = __expf(sv1 - newm);
            float rs   = p0 + p1;
            rs += __shfl_xor(rs, 1, 32);
            rs += __shfl_xor(rs, 2, 32);
            rs += __shfl_xor(rs, 4, 32);
            rs += __shfl_xor(rs, 8, 32);
            lrun[r] = lrun[r] * corr + rs;
            mrun[r] = newm;
            o0[r] = o0[r] * corr;
            o1[r] = o1[r] * corr;
            int prow = r + rhalf;
            Ps[prow * 32 + l15]      = f2bf(p0);
            Ps[prow * 32 + 16 + l15] = f2bf(p1);
        }
        __syncthreads();

        // O += P x V : P is A(16x32 keys); V is B(n=dim, K=key) via Vt rows
        const unsigned short* pr = &Ps[l15 * 32];
        v16bf pa = join16(*(const ushort8_t*)(pr + half * 16),
                          *(const ushort8_t*)(pr + half * 16 + 8));
        const unsigned short* v0r = &Vt[l15 * 32 + half * 16];
        const unsigned short* v1r = &Vt[(16 + l15) * 32 + half * 16];
        v16bf vb0 = join16(*(const ushort8_t*)(v0r), *(const ushort8_t*)(v0r + 8));
        v16bf vb1 = join16(*(const ushort8_t*)(v1r), *(const ushort8_t*)(v1r + 8));
        o0 = __builtin_amdgcn_wmma_f32_16x16x32_bf16(
                 false, pa, false, vb0, (short)0, o0, false, false);
        o1 = __builtin_amdgcn_wmma_f32_16x16x32_bf16(
                 false, pa, false, vb1, (short)0, o1, false, false);
        __syncthreads();
    }

#pragma unroll
    for (int r = 0; r < 8; ++r) {
        float inv = 1.0f / lrun[r];
        int row = bm + r + rhalf;
        O[(size_t)row * DM + hd0 + l15]      = f2bf(o0[r] * inv);
        O[(size_t)row * DM + hd0 + 16 + l15] = f2bf(o1[r] * inv);
    }
}

// ---------------------------------------------------------------------------
// LayerNorm over D=256: one wave per row; f32 in (h), bf16 out (GEMM A-side).
// ---------------------------------------------------------------------------
__global__ __launch_bounds__(256)
void layernorm256(const float* __restrict__ X, const float* __restrict__ s,
                  const float* __restrict__ b, unsigned short* __restrict__ Y)
{
    const int wid = threadIdx.x >> 5, lane = threadIdx.x & 31;
    const int row = blockIdx.x * 8 + wid;
    const float* xr = X + (size_t)row * DM;
    float vals[8], sum = 0.f;
#pragma unroll
    for (int j = 0; j < 8; ++j) { vals[j] = xr[lane + j * 32]; sum += vals[j]; }
#pragma unroll
    for (int off = 16; off >= 1; off >>= 1) sum += __shfl_xor(sum, off, 32);
    float mean = sum * (1.0f / DM);
    float var = 0.f;
#pragma unroll
    for (int j = 0; j < 8; ++j) { float d = vals[j] - mean; var += d * d; }
#pragma unroll
    for (int off = 16; off >= 1; off >>= 1) var += __shfl_xor(var, off, 32);
    float inv = rsqrtf(var * (1.0f / DM) + 1e-5f);
    unsigned short* yr = Y + (size_t)row * DM;
#pragma unroll
    for (int j = 0; j < 8; ++j) {
        int d = lane + j * 32;
        yr[d] = f2bf((vals[j] - mean) * inv * s[d] + b[d]);
    }
}

// x[b,c,h,w] (f32) -> xt[b*1600+n, c] (bf16)  (im2col for the 1x1 conv GEMM)
__global__ void transpose_x(const float* __restrict__ x,
                            unsigned short* __restrict__ xt)
{
    int idx = blockIdx.x * 256 + threadIdx.x;     // over b,c,n (source order)
    int n = idx % SEQ;
    int rc = idx / SEQ;
    int c = rc % CIN;
    int b = rc / CIN;
    xt[(size_t)(b * SEQ + n) * CIN + c] = f2bf(x[idx]);
}

// src[(b*1600+h*40+w)*(3*C) + a*C + cc] -> dst[b,a,h,w,cc] (flat)
__global__ void scatter_head(const float* __restrict__ src, float* __restrict__ dst,
                             int C)
{
    int idx = blockIdx.x * 256 + threadIdx.x;
    int cc = idx % C; int t = idx / C;
    int ww = t % HW; t /= HW;
    int hh = t % HW; t /= HW;
    int a = t % NAC; int b = t / NAC;
    dst[idx] = src[(size_t)(b * SEQ + hh * HW + ww) * (NAC * C) + a * C + cc];
}

// ---------------------------------------------------------------------------
extern "C" void kernel_launch(void* const* d_in, const int* in_sizes, int n_in,
                              void* d_out, int out_size, void* d_ws, size_t ws_size,
                              hipStream_t stream)
{
    const float* x      = (const float*)d_in[0];
    const float* conv_w = (const float*)d_in[1];
    const float* conv_b = (const float*)d_in[2];
    const float* pos    = (const float*)d_in[3];
    const float* qw = (const float*)d_in[4],  *qb = (const float*)d_in[5];
    const float* kw = (const float*)d_in[6],  *kb = (const float*)d_in[7];
    const float* vw = (const float*)d_in[8],  *vb = (const float*)d_in[9];
    const float* pw = (const float*)d_in[10], *pb = (const float*)d_in[11];
    const float* ln1s = (const float*)d_in[12], *ln1b = (const float*)d_in[13];
    const float* ln2s = (const float*)d_in[14], *ln2b = (const float*)d_in[15];
    const float* f1w = (const float*)d_in[16], *f1b = (const float*)d_in[17];
    const float* f2w = (const float*)d_in[18], *f2b = (const float*)d_in[19];
    const float* cls_w1 = (const float*)d_in[20], *cls_b1 = (const float*)d_in[21];
    const float* cls_w2 = (const float*)d_in[22], *cls_b2 = (const float*)d_in[23];
    const float* box_w1 = (const float*)d_in[24], *box_b1 = (const float*)d_in[25];
    const float* box_w2 = (const float*)d_in[26], *box_b2 = (const float*)d_in[27];
    const float* obj_w1 = (const float*)d_in[28], *obj_b1 = (const float*)d_in[29];
    const float* obj_w2 = (const float*)d_in[30], *obj_b2 = (const float*)d_in[31];

    // workspace layout: f32 region, then bf16 region (~60 MB total)
    float* ws = (float*)d_ws;
    const size_t MD = (size_t)MROWS * DM;          // 1,638,400
    float* h    = ws;
    float* clsb = h + MD;                          // 6400*240
    float* boxb = clsb + (size_t)MROWS * (NAC * NCLS);
    float* objb = boxb + (size_t)MROWS * (NAC * 4);
    unsigned short* hb = (unsigned short*)(objb + (size_t)MROWS * NAC);
    unsigned short* y  = hb + MD;
    unsigned short* q  = y + MD;
    unsigned short* k  = q + MD;
    unsigned short* v  = k + MD;
    unsigned short* o  = v + MD;
    unsigned short* t  = o + MD;                   // MROWS*FF (xt aliases this)
    unsigned short* xt = t;                        // MROWS*CIN fits inside t
    unsigned short* wscr = t + (size_t)MROWS * FF; // up to 2048*256 bf16

    auto gemm = [&](const unsigned short* A, const float* W, const float* bias,
                    const float* posp, const float* resid,
                    float* Cf, unsigned short* Cb, int K, int N, int relu) {
        int nblk = (N + 63) / 64, N64 = nblk * 64;
        convert_weight<<<(N64 * K) / 256, 256, 0, stream>>>(W, wscr, K, N);
        dim3 grid(nblk, MROWS / 64);
        gemm_bf16_wmma<<<grid, 128, 0, stream>>>(A, wscr, bias, posp, resid,
                                                 Cf, Cb, MROWS, K, N, relu);
    };

    // 1x1 conv projection + pos encoding
    transpose_x<<<(BATCH * CIN * SEQ) / 256, 256, 0, stream>>>(x, xt);
    gemm(xt, conv_w, conv_b, pos, nullptr, h, hb, CIN, DM, 0);

    for (int i = 0; i < NL; ++i) {
        layernorm256<<<MROWS / 8, 256, 0, stream>>>(h, ln1s + i * DM, ln1b + i * DM, y);
        gemm(y, qw + (size_t)i * DM * DM, qb + i * DM, nullptr, nullptr, nullptr, q, DM, DM, 0);
        gemm(y, kw + (size_t)i * DM * DM, kb + i * DM, nullptr, nullptr, nullptr, k, DM, DM, 0);
        gemm(y, vw + (size_t)i * DM * DM, vb + i * DM, nullptr, nullptr, nullptr, v, DM, DM, 0);
        attention_wmma<<<dim3(SEQ / 16, NH, BATCH), 32, 0, stream>>>(q, k, v, o);
        gemm(o, pw + (size_t)i * DM * DM, pb + i * DM, nullptr, h, h, hb, DM, DM, 0);
        layernorm256<<<MROWS / 8, 256, 0, stream>>>(h, ln2s + i * DM, ln2b + i * DM, y);
        gemm(y, f1w + (size_t)i * DM * FF, f1b + i * FF, nullptr, nullptr, nullptr, t, DM, FF, 1);
        gemm(t, f2w + (size_t)i * FF * DM, f2b + i * DM, nullptr, h, h, hb, FF, DM, 0);
    }

    // prediction heads (A-side reads bf16 shadow hb)
    gemm(hb, cls_w1, cls_b1, nullptr, nullptr, nullptr, y, DM, DM, 1);
    gemm(y, cls_w2, cls_b2, nullptr, nullptr, clsb, nullptr, DM, NAC * NCLS, 0);
    gemm(hb, box_w1, box_b1, nullptr, nullptr, nullptr, y, DM, DM, 1);
    gemm(y, box_w2, box_b2, nullptr, nullptr, boxb, nullptr, DM, NAC * 4, 0);
    gemm(hb, obj_w1, obj_b1, nullptr, nullptr, nullptr, y, DM, DM, 1);
    gemm(y, obj_w2, obj_b2, nullptr, nullptr, objb, nullptr, DM, NAC, 0);

    // final layout: cls [B,3,40,40,80] ++ box [B,3,40,40,4] ++ obj [B,3,40,40]
    float* out = (float*)d_out;
    const int cls_n = BATCH * NAC * SEQ * NCLS;    // 1,536,000
    const int box_n = BATCH * NAC * SEQ * 4;       //    76,800
    const int obj_n = BATCH * NAC * SEQ;           //    19,200
    scatter_head<<<cls_n / 256, 256, 0, stream>>>(clsb, out, NCLS);
    scatter_head<<<box_n / 256, 256, 0, stream>>>(boxb, out + cls_n, 4);
    scatter_head<<<obj_n / 256, 256, 0, stream>>>(objb, out + cls_n + box_n, 1);
}
